// ElementwiseSparsity_49486613185023
// MI455X (gfx1250) — compile-verified, matrix operand order checked by reading, not run on hardware
//
#include <hip/hip_runtime.h>
#include <hip/hip_bf16.h>

// Problem constants (match reference)
#define BB       8
#define MODEL_D  1024
#define HIGH_D   2048
#define TT       4096
#define KEEP     64

// GEMM tiling
#define BM   128
#define BN   128
#define BKk  32
#define BKP  40          // padded K stride in halves (80B rows -> 16B-aligned chunks)
#define NTHR 256         // 8 waves of 32

typedef __attribute__((ext_vector_type(16))) __bf16   v16bf;
typedef __attribute__((ext_vector_type(8)))  float    v8f;
typedef __attribute__((ext_vector_type(4)))  float    f32x4;
typedef __attribute__((ext_vector_type(4)))  unsigned u32x4;
typedef __attribute__((ext_vector_type(2)))  unsigned u32x2;

union FragU { u32x4 q[2]; v16bf v; };

static __device__ __forceinline__ unsigned short f2bf(float f) {
    unsigned u = __float_as_uint(f);
    unsigned r = u + 0x7FFFu + ((u >> 16) & 1u);   // round-to-nearest-even
    return (unsigned short)(r >> 16);
}
static __device__ __forceinline__ unsigned pack2(float lo, float hi) {
    return (unsigned)f2bf(lo) | ((unsigned)f2bf(hi) << 16);
}

// Out[R,T] = W[R,K] @ In[K,T] + bias[R], per batch (blockIdx.z), f32 in/out,
// bf16 WMMA math with f32 accumulation. All dims multiples of tile sizes here.
__global__ __launch_bounds__(NTHR)
void gemm_wmma_bf16(const float* __restrict__ W, const float* __restrict__ In,
                    const float* __restrict__ bias, float* __restrict__ Out,
                    int R, int K, int T) {
    __shared__ __align__(16) unsigned short lds_a[BM * BKP]; // [row][k] k-contiguous
    __shared__ __align__(16) unsigned short lds_b[BN * BKP]; // [col][k] (B transposed)

    const int tid  = threadIdx.x;
    const int lane = tid & 31;
    const int hi   = lane >> 4;          // lane half selects K sub-block
    const int l15  = lane & 15;
    const int wave = tid >> 5;
    const int wm   = wave & 3;           // 4 waves along M
    const int wn   = wave >> 2;          // 2 waves along N

    const int col0 = blockIdx.x * BN;    // along T
    const int row0 = blockIdx.y * BM;    // along R
    const size_t z = blockIdx.z;
    In  += z * (size_t)K * T;
    Out += z * (size_t)R * T;

    v8f acc[2][4];
#pragma unroll
    for (int i = 0; i < 2; ++i)
#pragma unroll
        for (int j = 0; j < 4; ++j) acc[i][j] = (v8f)0.0f;

    for (int k0 = 0; k0 < K; k0 += BKk) {
        // ---- stage A tile: W[row0+r][k0+c..c+3] -> lds_a[r][c..c+3], b128 loads ----
#pragma unroll
        for (int i = tid; i < BM * BKk / 4; i += NTHR) {   // 4 iterations
            int r  = i >> 3;              // 8 float4 per 32-wide row
            int c4 = (i & 7) * 4;
            f32x4 w = *(const f32x4*)&W[(size_t)(row0 + r) * K + (k0 + c4)];
            u32x2 p;
            p.x = pack2(w.x, w.y);
            p.y = pack2(w.z, w.w);
            *(u32x2*)&lds_a[r * BKP + c4] = p;             // one ds_store_b64
        }
        // ---- stage B tile transposed: In[k0+kk][col0+n..n+3] -> lds_b[n..n+3][kk] ----
#pragma unroll
        for (int i = tid; i < BKk * BN / 4; i += NTHR) {   // 4 iterations
            int kk = i >> 5;              // 32 float4 per 128-wide row
            int n4 = (i & 31) * 4;
            f32x4 v = *(const f32x4*)&In[(size_t)(k0 + kk) * T + (col0 + n4)];
            lds_b[(n4 + 0) * BKP + kk] = f2bf(v.x);
            lds_b[(n4 + 1) * BKP + kk] = f2bf(v.y);
            lds_b[(n4 + 2) * BKP + kk] = f2bf(v.z);
            lds_b[(n4 + 3) * BKP + kk] = f2bf(v.w);
        }
        // hint-prefetch next K tile (global_prefetch_b8 on gfx1250)
        if (k0 + BKk < K) {
            __builtin_prefetch(&W[(size_t)(row0 + (tid >> 1)) * K + (k0 + BKk)], 0, 1);
            __builtin_prefetch(&In[(size_t)(k0 + BKk + (tid >> 4)) * T + col0], 0, 1);
        }
        __syncthreads();

        // ---- load fragments (two 16B chunks per lane: K=[8hi,8hi+8),[16+8hi,+8)) ----
        FragU a[2], b[4];
#pragma unroll
        for (int fm = 0; fm < 2; ++fm) {
            const unsigned short* p = &lds_a[(wm * 32 + fm * 16 + l15) * BKP];
            a[fm].q[0] = *(const u32x4*)(p + 8 * hi);
            a[fm].q[1] = *(const u32x4*)(p + 16 + 8 * hi);
        }
#pragma unroll
        for (int fn = 0; fn < 4; ++fn) {
            const unsigned short* p = &lds_b[(wn * 64 + fn * 16 + l15) * BKP];
            b[fn].q[0] = *(const u32x4*)(p + 8 * hi);
            b[fn].q[1] = *(const u32x4*)(p + 16 + 8 * hi);
        }

        // ---- 8 WMMAs per wave per K-step ----
#pragma unroll
        for (int fm = 0; fm < 2; ++fm)
#pragma unroll
            for (int fn = 0; fn < 4; ++fn)
                acc[fm][fn] = __builtin_amdgcn_wmma_f32_16x16x32_bf16(
                    false, a[fm].v, false, b[fn].v,
                    (short)0, acc[fm][fn], false, false);

        __syncthreads();
    }

    // ---- epilogue: bias add + store (C/D layout: VGPR v -> row v+8*hi, lane -> col) ----
#pragma unroll
    for (int fm = 0; fm < 2; ++fm) {
#pragma unroll
        for (int v = 0; v < 8; ++v) {
            int grow = row0 + wm * 32 + fm * 16 + v + 8 * hi;
            float bv = bias[grow];
#pragma unroll
            for (int fn = 0; fn < 4; ++fn) {
                int gcol = col0 + wn * 64 + fn * 16 + l15;
                Out[(size_t)grow * T + gcol] = acc[fm][fn][v] + bv;
            }
        }
    }
}

// Exact in-place top-KEEP per row of TT floats via 32-pass MSB radix select in LDS.
static __device__ __forceinline__ unsigned fkey(float f) {
    unsigned u = __float_as_uint(f);
    return (u & 0x80000000u) ? ~u : (u | 0x80000000u);   // monotone order-preserving
}

__global__ __launch_bounds__(NTHR)
void topk_inplace(float* __restrict__ S) {
    __shared__ __align__(16) float vals[TT];
    __shared__ unsigned cnt, tieCnt;

    float* row = S + (size_t)blockIdx.x * TT;
    const int tid = threadIdx.x;

#pragma unroll
    for (int i = tid; i < TT / 4; i += NTHR)
        ((f32x4*)vals)[i] = ((const f32x4*)row)[i];
    __syncthreads();

    unsigned pref = 0, gt = 0;
    for (int bit = 31; bit >= 0; --bit) {
        if (tid == 0) cnt = 0;
        __syncthreads();
        unsigned maskHi = (bit >= 31) ? 0u : (0xFFFFFFFFu << (bit + 1));
        unsigned local = 0;
        for (int i = tid; i < TT; i += NTHR) {
            unsigned k = fkey(vals[i]);
            if (((k & maskHi) == pref) && ((k >> bit) & 1u)) ++local;
        }
        if (local) atomicAdd(&cnt, local);
        __syncthreads();
        unsigned c = cnt;
        __syncthreads();
        if (gt + c >= KEEP) pref |= (1u << bit);
        else                gt  += c;
    }
    // pref == key of KEEP-th largest; gt == #elements strictly greater
    unsigned tiesNeeded = KEEP - gt;
    if (tid == 0) tieCnt = 0;
    __syncthreads();
    for (int i = tid; i < TT / 4; i += NTHR) {
        f32x4 v = ((const f32x4*)vals)[i];
        f32x4 o;
#pragma unroll
        for (int j = 0; j < 4; ++j) {
            float f = v[j];
            unsigned k = fkey(f);
            float r = 0.0f;
            if (k > pref) r = f;
            else if (k == pref) {
                unsigned s = atomicAdd(&tieCnt, 1u);
                if (s < tiesNeeded) r = f;
            }
            o[j] = r;
        }
        ((f32x4*)row)[i] = o;
    }
}

extern "C" void kernel_launch(void* const* d_in, const int* in_sizes, int n_in,
                              void* d_out, int out_size, void* d_ws, size_t ws_size,
                              hipStream_t stream) {
    const float* x   = (const float*)d_in[0];  // [B, C=1024, T]
    const float* W_e = (const float*)d_in[1];  // [D=2048, C]
    const float* b_e = (const float*)d_in[2];  // [D]
    const float* W_c = (const float*)d_in[3];  // [M=1024, D]
    const float* b_c = (const float*)d_in[4];  // [M]

    float* out    = (float*)d_out;                               // [B, M, T]
    float* sparse = out + (size_t)BB * MODEL_D * TT;             // [B, D, T] (also output)

    // 1) expand: Y[b] = W_e @ x[b] + b_e  -> written straight into sparse slot
    gemm_wmma_bf16<<<dim3(TT / BN, HIGH_D / BM, BB), NTHR, 0, stream>>>(
        W_e, x, b_e, sparse, HIGH_D, MODEL_D, TT);

    // 2) exact top-KEEP along T, in place, per (b,d) row
    topk_inplace<<<dim3(BB * HIGH_D), NTHR, 0, stream>>>(sparse);

    // 3) contract: out[b] = W_c @ sparse[b] + b_c
    gemm_wmma_bf16<<<dim3(TT / BN, MODEL_D / BM, BB), NTHR, 0, stream>>>(
        W_c, sparse, b_c, out, MODEL_D, HIGH_D, TT);
}
